// EGNN_65532611002593
// MI455X (gfx1250) — matrix operand order repaired and checked by reading, not compiled, not run on hardware
//
#include <hip/hip_runtime.h>

#define NODES   10000
#define NEDGES  160000
#define NGRAPHS 64
#define EMB     128
#define MDIM    32
#define FF      32
#define EDGE_IN 321       // 2*FF + 1 + 2*EMB
#define NK      5
#define MLP     256
#define EPSLN   1e-5f

// ---- GEMM tilings (K padded to mult of 32, N to mult of 16) ----
#define KT1 11            // edge GEMM1: K 321 -> 352
#define NT1 41            // edge GEMM1: N 642 -> 656
#define KA1 352
#define HID 642
#define KT2 21            // edge GEMM2: K 642 -> 672
#define KA2 672
#define NT2 2             // N = 32

#define ETILES 4          // edge M-tiles (16 edges each) per wave: B reuse x4

#define NKT1 5            // node GEMM1: K 160
#define NNT1 16           // node GEMM1: N 256
#define NKT2 8            // node GEMM2: K 256
#define NNT2 8            // node GEMM2: N 128

#define FKT1 24           // final GEMM1: K 768
#define FNT1 16           // N 256
#define FKT2 8            // K 256
#define FNT2 16           // N 256

typedef _Float16 half_t;
typedef __attribute__((ext_vector_type(16))) _Float16 v16h;
typedef __attribute__((ext_vector_type(8)))  _Float16 v8h;
typedef __attribute__((ext_vector_type(8)))  float    v8f;

__device__ __forceinline__ v16h cat8(v8h lo, v8h hi) {
  v16h r;
#pragma unroll
  for (int i = 0; i < 8; ++i) { r[i] = lo[i]; r[i + 8] = hi[i]; }
  return r;
}

// A-matrix fragment from LDS, per CDNA5 16-bit A layout:
// lane L (M = L&15): halves 0..7 = K[ko..ko+7], halves 8..15 = K[16+ko..23+ko], ko = 8*(L>=16)
__device__ __forceinline__ v16h load_a(const half_t* A, int lane, int stride, int kk) {
  const half_t* p = A + (lane & 15) * stride + kk + ((lane >> 4) << 3);
  return cat8(*(const v8h*)p, *(const v8h*)(p + 16));
}

// B-matrix fragment from repacked global weights: lane-contiguous 32B
__device__ __forceinline__ v16h load_b(const half_t* B, int lane) {
  const half_t* p = B + lane * 16;
  return cat8(*(const v8h*)p, *(const v8h*)(p + 8));
}

__device__ __forceinline__ v8f wmma16(v16h a, v16h b, v8f c) {
  return __builtin_amdgcn_wmma_f32_16x16x32_f16(false, a, false, b, (short)0, c, false, false);
}

__device__ __forceinline__ float silu(float x) { return x * (1.f / (1.f + __expf(-x))); }

// ---------------- weight repack: f32 (K x N) -> f16 WMMA-B tiles -------------
// tileIdx = nt*KT + kt ; within tile: element (L, j): K = kt*32 + 8*(L>=16) + (j&7) + 16*(j>=8),
// N = nt*16 + (L&15); stored at tileIdx*512 + L*16 + j. OOB -> 0 (true zeros so
// activation-pad garbage can never produce NaN via 0*Inf).
__global__ void repack_kernel(const float* __restrict__ W, half_t* __restrict__ out,
                              int K, int N, int KT, int NT) {
  int t = blockIdx.x * blockDim.x + threadIdx.x;
  int total = KT * NT * 512;
  if (t >= total) return;
  int tile = t >> 9;
  int r = t & 511;
  int L = r >> 4, j = r & 15;
  int nt = tile / KT, kt = tile % KT;
  int k = kt * 32 + ((L >> 4) << 3) + (j & 7) + ((j >> 3) << 4);
  int n = nt * 16 + (L & 15);
  float v = (k < K && n < N) ? W[(size_t)k * N + n] : 0.f;
  out[t] = (half_t)v;
}

// ---------------- misc small kernels ----------------------------------------
__global__ void zero_kernel(float* p, int n) {
  int t = blockIdx.x * blockDim.x + threadIdx.x;
  if (t < n) p[t] = 0.f;
}

__global__ void embed_kernel(const int* __restrict__ atomids,
                             const float* __restrict__ emb_w,
                             float* __restrict__ feats0) {
  int t = blockIdx.x * blockDim.x + threadIdx.x;
  if (t >= NODES * EMB) return;
  int i = t >> 7, c = t & 127;
  feats0[t] = emb_w[atomids[i] * EMB + c];
}

__global__ void d2_kernel(const int* __restrict__ ei,
                          const float* __restrict__ coords,
                          float* __restrict__ d2buf) {
  int e = blockIdx.x * blockDim.x + threadIdx.x;
  if (e >= NEDGES) return;
  int si = ei[e], di = ei[NEDGES + e];
  float dx = coords[si * 3 + 0] - coords[di * 3 + 0];
  float dy = coords[si * 3 + 1] - coords[di * 3 + 1];
  float dz = coords[si * 3 + 2] - coords[di * 3 + 2];
  d2buf[e] = dx * dx + dy * dy + dz * dz;
}

// ---------------- edge MLP: 64 edges (4 M-tiles) per wave --------------------
__global__ __launch_bounds__(32)
void edge_kernel(const int* __restrict__ ei, const float* __restrict__ d2buf,
                 const float* __restrict__ feats,
                 const half_t* __restrict__ W1, const half_t* __restrict__ W2,
                 const float* __restrict__ b1, const float* __restrict__ b2,
                 const float* __restrict__ lg, const float* __restrict__ lb,
                 float* __restrict__ sbuf, float* __restrict__ cntbuf) {
  __shared__ __align__(16) half_t Ash[ETILES * 16 * KA1];
  __shared__ __align__(16) half_t Hsh[ETILES * 16 * KA2];
  __shared__ __align__(16) float  Fsh[ETILES * 16 * MDIM];
  int lane = threadIdx.x;
  int e0 = blockIdx.x * (16 * ETILES);

  // phase 1: build 4 A tiles [feats[dst] | feats[src] | sin | cos | d2 | 0-pad]
  for (int t = 0; t < ETILES; ++t) {
    half_t* At = Ash + t * 16 * KA1;
    half_t* Ht = Hsh + t * 16 * KA2;
    for (int r = 0; r < 16; ++r) {
      int e = e0 + t * 16 + r;
      int si = ei[e];
      int di = ei[NEDGES + e];
      const float* fs = feats + (size_t)si * EMB;
      const float* fd = feats + (size_t)di * EMB;
      for (int c = lane; c < EMB; c += 32) {
        At[r * KA1 + c]       = (half_t)fd[c];
        At[r * KA1 + EMB + c] = (half_t)fs[c];
      }
      float d2 = d2buf[e];
      float xs = d2 * exp2f(-(float)lane);        // lane = Fourier scale index (FF==32)
      At[r * KA1 + 2 * EMB + lane]      = (half_t)__sinf(xs);
      At[r * KA1 + 2 * EMB + FF + lane] = (half_t)__cosf(xs);
      if (lane == 0) At[r * KA1 + 320] = (half_t)d2;
      int c = EDGE_IN + lane;
      if (c < KA1) At[r * KA1 + c] = (half_t)0.f;   // K pad
      int hc = HID + lane;
      if (hc < KA2) Ht[r * KA2 + hc] = (half_t)0.f; // hidden K pad
    }
  }
  __syncthreads();

  int colL = lane & 15;
  int rowBase = (lane >> 4) << 3;

  // GEMM1: 4x(16x352) @ 352x656 -> silu -> hidden (f16 in LDS).
  // One B fragment feeds 4 independent WMMAs (4x B reuse, 4-way matrix ILP).
  for (int nt = 0; nt < NT1; ++nt) {
    v8f acc[ETILES] = {};
    const half_t* Bt = W1 + (size_t)(nt * KT1) * 512;
    for (int kt = 0; kt < KT1; ++kt) {
      v16h b = load_b(Bt + kt * 512, lane);
      if (kt + 1 < KT1) __builtin_prefetch(Bt + (kt + 1) * 512 + lane * 16, 0, 3);
#pragma unroll
      for (int t = 0; t < ETILES; ++t)
        acc[t] = wmma16(load_a(Ash + t * 16 * KA1, lane, KA1, kt * 32), b, acc[t]);
    }
    int col = nt * 16 + colL;
    float bias = (col < HID) ? b1[col] : 0.f;
#pragma unroll
    for (int t = 0; t < ETILES; ++t) {
#pragma unroll
      for (int r = 0; r < 8; ++r) {
        float sv = silu(acc[t][r] + bias);
        if (col < HID) Hsh[t * 16 * KA2 + (rowBase + r) * KA2 + col] = (half_t)sv;
      }
    }
  }
  __syncthreads();

  // GEMM2: 4x(16x672) @ 672x32 -> silu -> f32 stage
  for (int nt = 0; nt < NT2; ++nt) {
    v8f acc[ETILES] = {};
    const half_t* Bt = W2 + (size_t)(nt * KT2) * 512;
    for (int kt = 0; kt < KT2; ++kt) {
      v16h b = load_b(Bt + kt * 512, lane);
#pragma unroll
      for (int t = 0; t < ETILES; ++t)
        acc[t] = wmma16(load_a(Hsh + t * 16 * KA2, lane, KA2, kt * 32), b, acc[t]);
    }
    int col = nt * 16 + colL;
    float bias = b2[col];
#pragma unroll
    for (int t = 0; t < ETILES; ++t) {
#pragma unroll
      for (int r = 0; r < 8; ++r)
        Fsh[t * 16 * MDIM + (rowBase + r) * MDIM + col] = silu(acc[t][r] + bias);
    }
  }
  __syncthreads();

  // per-edge LayerNorm(en) + atomic segment-sum scatter on dst
  for (int t = 0; t < ETILES; ++t) {
    const float* F = Fsh + t * 16 * MDIM;
    int row = lane & 15;
    int hsel = lane >> 4;
    float mean = 0.f;
    for (int c = 0; c < MDIM; ++c) mean += F[row * MDIM + c];
    mean *= (1.f / MDIM);
    float var = 0.f;
    for (int c = 0; c < MDIM; ++c) {
      float d = F[row * MDIM + c] - mean;
      var += d * d;
    }
    var *= (1.f / MDIM);
    float rs = rsqrtf(var + EPSLN);
    int e = e0 + t * 16 + row;
    int di = ei[NEDGES + e];
    for (int c = hsel * 16; c < hsel * 16 + 16; ++c) {
      float y = (F[row * MDIM + c] - mean) * rs * lg[c] + lb[c];
      atomicAdd(&sbuf[(size_t)di * MDIM + c], y);
    }
    if (hsel == 0) atomicAdd(&cntbuf[di], 1.f);
  }
}

// ---------------- node MLP: 16 nodes per wave --------------------------------
__global__ __launch_bounds__(32)
void node_kernel(const float* __restrict__ fin, float* __restrict__ fout,
                 const float* __restrict__ sbuf, const float* __restrict__ cntbuf,
                 const float* __restrict__ eng, const float* __restrict__ enb,
                 const float* __restrict__ n1g, const float* __restrict__ n1b,
                 const half_t* __restrict__ W1, const float* __restrict__ b1,
                 const half_t* __restrict__ W2, const float* __restrict__ b2,
                 const float* __restrict__ n2g, const float* __restrict__ n2b) {
  __shared__ __align__(16) half_t Ash[16 * 160];
  __shared__ __align__(16) half_t Hsh[16 * 256];
  __shared__ __align__(16) float  Fsh[16 * 128];
  int lane = threadIdx.x;
  int i0 = blockIdx.x * 16;
  int row = lane & 15, hsel = lane >> 4;
  int i = i0 + row;

  // build A = [ LN(feats, nn1) (128) | LN(s/max(cnt,1), en) (32) ]
  {
    const float* f = fin + (size_t)i * EMB;
    float mean = 0.f;
    for (int c = 0; c < EMB; ++c) mean += f[c];
    mean *= (1.f / EMB);
    float var = 0.f;
    for (int c = 0; c < EMB; ++c) { float d = f[c] - mean; var += d * d; }
    var *= (1.f / EMB);
    float rs = rsqrtf(var + EPSLN);
    for (int c = hsel * 64; c < hsel * 64 + 64; ++c)
      Ash[row * 160 + c] = (half_t)((f[c] - mean) * rs * n1g[c] + n1b[c]);

    float cn = cntbuf[i]; cn = cn > 1.f ? cn : 1.f;
    float inv = 1.f / cn;
    const float* sp = sbuf + (size_t)i * MDIM;
    float m2 = 0.f;
    for (int c = 0; c < MDIM; ++c) m2 += sp[c] * inv;
    m2 *= (1.f / MDIM);
    float v2 = 0.f;
    for (int c = 0; c < MDIM; ++c) { float d = sp[c] * inv - m2; v2 += d * d; }
    v2 *= (1.f / MDIM);
    float rs2 = rsqrtf(v2 + EPSLN);
    for (int c = hsel * 16; c < hsel * 16 + 16; ++c)
      Ash[row * 160 + EMB + c] = (half_t)((sp[c] * inv - m2) * rs2 * eng[c] + enb[c]);
  }
  __syncthreads();

  int colL = lane & 15;
  int rowBase = (lane >> 4) << 3;

  // GEMM1: 16x160 @ 160x256, silu
  for (int nt = 0; nt < NNT1; ++nt) {
    v8f acc = {};
    const half_t* Bt = W1 + (size_t)(nt * NKT1) * 512;
    for (int kt = 0; kt < NKT1; ++kt)
      acc = wmma16(load_a(Ash, lane, 160, kt * 32), load_b(Bt + kt * 512, lane), acc);
    int col = nt * 16 + colL;
    float bias = b1[col];
#pragma unroll
    for (int r = 0; r < 8; ++r)
      Hsh[(rowBase + r) * 256 + col] = (half_t)silu(acc[r] + bias);
  }
  __syncthreads();

  // GEMM2: 16x256 @ 256x128 (no activation)
  for (int nt = 0; nt < NNT2; ++nt) {
    v8f acc = {};
    const half_t* Bt = W2 + (size_t)(nt * NKT2) * 512;
    for (int kt = 0; kt < NKT2; ++kt)
      acc = wmma16(load_a(Hsh, lane, 256, kt * 32), load_b(Bt + kt * 512, lane), acc);
    int col = nt * 16 + colL;
    float bias = b2[col];
#pragma unroll
    for (int r = 0; r < 8; ++r)
      Fsh[(rowBase + r) * 128 + col] = acc[r] + bias;
  }
  __syncthreads();

  // LN(nn2) + residual
  {
    float mean = 0.f;
    for (int c = 0; c < EMB; ++c) mean += Fsh[row * 128 + c];
    mean *= (1.f / EMB);
    float var = 0.f;
    for (int c = 0; c < EMB; ++c) { float d = Fsh[row * 128 + c] - mean; var += d * d; }
    var *= (1.f / EMB);
    float rs = rsqrtf(var + EPSLN);
    for (int c = hsel * 64; c < hsel * 64 + 64; ++c) {
      float hn = (Fsh[row * 128 + c] - mean) * rs * n2g[c] + n2b[c];
      fout[(size_t)i * EMB + c] = fin[(size_t)i * EMB + c] + hn;
    }
  }
}

// ---------------- final MLP + graph pool: 16 nodes per wave ------------------
__global__ __launch_bounds__(32)
void final_kernel(const float* __restrict__ featsAll, const int* __restrict__ batch,
                  const half_t* __restrict__ W1, const float* __restrict__ b1,
                  const half_t* __restrict__ W2, const float* __restrict__ b2,
                  const half_t* __restrict__ W3, const float* __restrict__ b3,
                  float* __restrict__ gs, float* __restrict__ gcnt) {
  __shared__ __align__(16) half_t Ash[16 * 768];
  __shared__ __align__(16) half_t Hsh[16 * 256];
  __shared__ __align__(16) half_t H2sh[16 * 256];
  int lane = threadIdx.x;
  int i0 = blockIdx.x * 16;

  // A = silu(concat(feat_list))  (768 = 6 x 128 slices)
  for (int r = 0; r < 16; ++r) {
    size_t ib = (size_t)(i0 + r) * EMB;
    for (int c = lane; c < 768; c += 32) {
      float v = featsAll[(size_t)(c >> 7) * NODES * EMB + ib + (c & 127)];
      Ash[r * 768 + c] = (half_t)silu(v);
    }
  }
  __syncthreads();

  int colL = lane & 15;
  int rowBase = (lane >> 4) << 3;

  // 768 -> 256, silu
  for (int nt = 0; nt < FNT1; ++nt) {
    v8f acc = {};
    const half_t* Bt = W1 + (size_t)(nt * FKT1) * 512;
    for (int kt = 0; kt < FKT1; ++kt)
      acc = wmma16(load_a(Ash, lane, 768, kt * 32), load_b(Bt + kt * 512, lane), acc);
    int col = nt * 16 + colL;
    float bias = b1[col];
#pragma unroll
    for (int r = 0; r < 8; ++r)
      Hsh[(rowBase + r) * 256 + col] = (half_t)silu(acc[r] + bias);
  }
  __syncthreads();

  // 256 -> 256, silu
  for (int nt = 0; nt < FNT2; ++nt) {
    v8f acc = {};
    const half_t* Bt = W2 + (size_t)(nt * FKT2) * 512;
    for (int kt = 0; kt < FKT2; ++kt)
      acc = wmma16(load_a(Hsh, lane, 256, kt * 32), load_b(Bt + kt * 512, lane), acc);
    int col = nt * 16 + colL;
    float bias = b2[col];
#pragma unroll
    for (int r = 0; r < 8; ++r)
      H2sh[(rowBase + r) * 256 + col] = (half_t)silu(acc[r] + bias);
  }
  __syncthreads();

  // 256 -> 256, then silu, atomic mean-pool into graph slots
  for (int nt = 0; nt < FNT2; ++nt) {
    v8f acc = {};
    const half_t* Bt = W3 + (size_t)(nt * FKT2) * 512;
    for (int kt = 0; kt < FKT2; ++kt)
      acc = wmma16(load_a(H2sh, lane, 256, kt * 32), load_b(Bt + kt * 512, lane), acc);
    int col = nt * 16 + colL;
    float bias = b3[col];
#pragma unroll
    for (int r = 0; r < 8; ++r) {
      int row = rowBase + r;
      float sv = silu(acc[r] + bias);
      int g = batch[i0 + row];
      atomicAdd(&gs[(size_t)g * MLP + col], sv);
    }
  }
  if (lane < 16) atomicAdd(&gcnt[batch[i0 + lane]], 1.f);
}

// ---------------- graph head (tiny, plain VALU) ------------------------------
__global__ __launch_bounds__(256)
void head_kernel(const float* __restrict__ gs, const float* __restrict__ gcnt,
                 const float* __restrict__ g1W, const float* __restrict__ g1b,
                 const float* __restrict__ g2W, const float* __restrict__ g2b,
                 const float* __restrict__ g3W, const float* __restrict__ g3b,
                 float* __restrict__ out) {
  __shared__ float h0[MLP], h1[MLP];
  int g = blockIdx.x, t = threadIdx.x;
  float c = gcnt[g]; c = c > 1.f ? c : 1.f;
  h0[t] = gs[(size_t)g * MLP + t] / c;
  __syncthreads();
  float a1 = g1b[t];
  for (int kx = 0; kx < MLP; ++kx) a1 += h0[kx] * g1W[kx * MLP + t];
  h1[t] = silu(a1);
  __syncthreads();
  float a2 = g2b[t];
  for (int kx = 0; kx < MLP; ++kx) a2 += h1[kx] * g2W[kx * MLP + t];
  __syncthreads();
  h0[t] = silu(a2);
  __syncthreads();
  if (t == 0) {
    float a3 = g3b[0];
    for (int kx = 0; kx < MLP; ++kx) a3 += h0[kx] * g3W[kx];
    out[g] = a3;
  }
}

// ---------------- launcher ---------------------------------------------------
extern "C" void kernel_launch(void* const* d_in, const int* in_sizes, int n_in,
                              void* d_out, int out_size, void* d_ws, size_t ws_size,
                              hipStream_t stream) {
  (void)in_sizes; (void)n_in; (void)out_size; (void)ws_size;
  const int*   atomids = (const int*)d_in[0];
  const float* coords  = (const float*)d_in[1];
  const int*   ei      = (const int*)d_in[2];
  const int*   batch   = (const int*)d_in[3];
  const float* emb_w   = (const float*)d_in[4];
  const float* eW1 = (const float*)d_in[5];
  const float* eb1 = (const float*)d_in[6];
  const float* eW2 = (const float*)d_in[7];
  const float* eb2 = (const float*)d_in[8];
  const float* en_g = (const float*)d_in[9];
  const float* en_b = (const float*)d_in[10];
  const float* nn1_g = (const float*)d_in[11];
  const float* nn1_b = (const float*)d_in[12];
  const float* nW1 = (const float*)d_in[13];
  const float* nb1 = (const float*)d_in[14];
  const float* nW2 = (const float*)d_in[15];
  const float* nb2 = (const float*)d_in[16];
  const float* nn2_g = (const float*)d_in[17];
  const float* nn2_b = (const float*)d_in[18];
  const float* f1W = (const float*)d_in[19];
  const float* f1b = (const float*)d_in[20];
  const float* f2W = (const float*)d_in[21];
  const float* f2b = (const float*)d_in[22];
  const float* f3W = (const float*)d_in[23];
  const float* f3b = (const float*)d_in[24];
  const float* g1W = (const float*)d_in[25];
  const float* g1b = (const float*)d_in[26];
  const float* g2W = (const float*)d_in[27];
  const float* g2b = (const float*)d_in[28];
  const float* g3W = (const float*)d_in[29];
  const float* g3b = (const float*)d_in[30];

  char* ws = (char*)d_ws;
  size_t off = 0;
  auto walloc = [&](size_t bytes) -> size_t {
    size_t o = off;
    off += (bytes + 255) & ~(size_t)255;
    return o;
  };
  float* featsAll = (float*)(ws + walloc((size_t)(NK + 1) * NODES * EMB * 4));
  float* d2buf    = (float*)(ws + walloc((size_t)NEDGES * 4));
  float* sbuf     = (float*)(ws + walloc((size_t)NODES * MDIM * 4));
  float* cntbuf   = (float*)(ws + walloc((size_t)NODES * 4));
  float* gs       = (float*)(ws + walloc((size_t)NGRAPHS * MLP * 4));
  float* gcnt     = (float*)(ws + walloc((size_t)NGRAPHS * 4));
  const size_t eW1sz = (size_t)KT1 * NT1 * 512;   // halves per layer
  const size_t eW2sz = (size_t)KT2 * NT2 * 512;
  const size_t nW1sz = (size_t)NKT1 * NNT1 * 512;
  const size_t nW2sz = (size_t)NKT2 * NNT2 * 512;
  half_t* weW1 = (half_t*)(ws + walloc((size_t)NK * eW1sz * 2));
  half_t* weW2 = (half_t*)(ws + walloc((size_t)NK * eW2sz * 2));
  half_t* wnW1 = (half_t*)(ws + walloc((size_t)NK * nW1sz * 2));
  half_t* wnW2 = (half_t*)(ws + walloc((size_t)NK * nW2sz * 2));
  half_t* wf1  = (half_t*)(ws + walloc((size_t)FKT1 * FNT1 * 512 * 2));
  half_t* wf2  = (half_t*)(ws + walloc((size_t)FKT2 * FNT2 * 512 * 2));
  half_t* wf3  = (half_t*)(ws + walloc((size_t)FKT2 * FNT2 * 512 * 2));

  auto repack = [&](const float* W, half_t* o, int K, int N, int KT, int NT) {
    int total = KT * NT * 512;
    repack_kernel<<<(total + 255) / 256, 256, 0, stream>>>(W, o, K, N, KT, NT);
  };
  for (int k = 0; k < NK; ++k) {
    repack(eW1 + (size_t)k * EDGE_IN * 2 * EDGE_IN, weW1 + k * eW1sz, EDGE_IN, 2 * EDGE_IN, KT1, NT1);
    repack(eW2 + (size_t)k * 2 * EDGE_IN * MDIM,    weW2 + k * eW2sz, 2 * EDGE_IN, MDIM, KT2, NT2);
    repack(nW1 + (size_t)k * (EMB + MDIM) * 2 * EMB, wnW1 + k * nW1sz, EMB + MDIM, 2 * EMB, NKT1, NNT1);
    repack(nW2 + (size_t)k * 2 * EMB * EMB,          wnW2 + k * nW2sz, 2 * EMB, EMB, NKT2, NNT2);
  }
  repack(f1W, wf1, (NK + 1) * EMB, MLP, FKT1, FNT1);
  repack(f2W, wf2, MLP, MLP, FKT2, FNT2);
  repack(f3W, wf3, MLP, MLP, FKT2, FNT2);

  embed_kernel<<<(NODES * EMB + 255) / 256, 256, 0, stream>>>(atomids, emb_w, featsAll);
  d2_kernel<<<(NEDGES + 255) / 256, 256, 0, stream>>>(ei, coords, d2buf);
  zero_kernel<<<(NGRAPHS * MLP + 255) / 256, 256, 0, stream>>>(gs, NGRAPHS * MLP);
  zero_kernel<<<1, 256, 0, stream>>>(gcnt, NGRAPHS);

  for (int k = 0; k < NK; ++k) {
    zero_kernel<<<(NODES * MDIM + 255) / 256, 256, 0, stream>>>(sbuf, NODES * MDIM);
    zero_kernel<<<(NODES + 255) / 256, 256, 0, stream>>>(cntbuf, NODES);
    const float* fin = featsAll + (size_t)k * NODES * EMB;
    float* fout = featsAll + (size_t)(k + 1) * NODES * EMB;
    edge_kernel<<<NEDGES / (16 * ETILES), 32, 0, stream>>>(
        ei, d2buf, fin, weW1 + k * eW1sz, weW2 + k * eW2sz,
        eb1 + (size_t)k * 2 * EDGE_IN, eb2 + (size_t)k * MDIM,
        en_g + (size_t)k * MDIM, en_b + (size_t)k * MDIM, sbuf, cntbuf);
    node_kernel<<<NODES / 16, 32, 0, stream>>>(
        fin, fout, sbuf, cntbuf,
        en_g + (size_t)k * MDIM, en_b + (size_t)k * MDIM,
        nn1_g + (size_t)k * EMB, nn1_b + (size_t)k * EMB,
        wnW1 + k * nW1sz, nb1 + (size_t)k * 2 * EMB,
        wnW2 + k * nW2sz, nb2 + (size_t)k * EMB,
        nn2_g + (size_t)k * EMB, nn2_b + (size_t)k * EMB);
  }

  final_kernel<<<NODES / 16, 32, 0, stream>>>(featsAll, batch, wf1, f1b, wf2, f2b, wf3, f3b, gs, gcnt);
  head_kernel<<<NGRAPHS, MLP, 0, stream>>>(gs, gcnt, g1W, g1b, g2W, g2b, g3W, g3b, (float*)d_out);
}
